// Seq2SeqAttention_90572270338210
// MI455X (gfx1250) — matrix operand order repaired and checked by reading, not compiled, not run on hardware
//
#include <hip/hip_runtime.h>
#include <hip/hip_bf16.h>

// ---------------------------------------------------------------------------
// Seq2Seq LSTM + Bahdanau attention forward on MI455X (gfx1250, wave32, WMMA)
// B=64, S=256, T=256, E=256, H=512, V_TGT=4000
// All GEMMs run on v_wmma_f32_16x16x32_bf16 (bf16 in, f32 accumulate).
//  - enc-side attention projection hoisted out of the decoder loop (once).
//  - GEMM: one wave = 1 N-tile x 4 M-tiles -> B fragment loaded once per
//    K-step, reused by 4 WMMAs (4x less weight traffic, 4x WMMA density).
// ---------------------------------------------------------------------------

#define B_   64
#define S_   256
#define T_   256
#define E_   256
#define H_   512
#define V_   4000
#define G4H_ (4 * H_)

typedef unsigned short bf16_t; // bf16 storage type (host-visible)
typedef __attribute__((ext_vector_type(16))) __bf16 v16bf;
typedef __attribute__((ext_vector_type(8)))  float  v8f;

// ---------------- bf16 helpers ----------------
__device__ __forceinline__ bf16_t f2bf(float x) {
  union { float f; unsigned u; } v; v.f = x;
  unsigned r = (v.u + 0x7FFFu + ((v.u >> 16) & 1u)) >> 16;
  return (bf16_t)r;
}
__device__ __forceinline__ __bf16 asbf(bf16_t h) {
  __bf16 b; __builtin_memcpy(&b, &h, 2); return b;
}

// ---------------- WMMA fragment loaders (wave32; contiguous K = 16*(lane/16) + s, same in A and B) -------
// A fragment: 16x32 bf16.  lanes 0..15 -> M=lane (K 0..15), lanes 16..31 ->
// M=lane-16 (K offset +16).  VGPR j holds K = khalf + 2*j.
// Per lane this is two contiguous 16B chunks -> compiler emits b128 loads.
__device__ __forceinline__ v16bf load_a_frag(const bf16_t* __restrict__ A,
                                             long ld, int k, int lane) {
  const bf16_t* p = A + (size_t)(lane & 15) * (size_t)ld + k + ((lane & 16) ? 16 : 0);
  v16bf a;
#pragma unroll
  for (int j = 0; j < 8; ++j) {
    int kk = j << 1;
    a[2 * j]     = asbf(p[kk]);
    a[2 * j + 1] = asbf(p[kk + 1]);
  }
  return a;
}
// B fragment: 32x16 bf16 where Bmat[k][n] = W[n][k] (row-major weight [N][K]).
// lanes 0..15 -> N=lane (K 0..15), lanes 16..31 -> same N, K offset +16.
// Per lane: 32 contiguous bytes -> two b128 loads.
__device__ __forceinline__ v16bf load_b_fragT(const bf16_t* __restrict__ W,
                                              long ld, int k, int lane) {
  const bf16_t* p = W + (size_t)(lane & 15) * (size_t)ld + k + ((lane & 16) ? 16 : 0);
  v16bf b;
#pragma unroll
  for (int j = 0; j < 8; ++j) {
    b[2 * j]     = asbf(p[2 * j]);
    b[2 * j + 1] = asbf(p[2 * j + 1]);
  }
  return b;
}

// ---------------- generic WMMA GEMM:  out = A1@W1.T (+ A2@W2.T) + bias -----
// One wave per {N-tile x 4 consecutive M-tiles}. B fragment loaded once per
// K-step and reused by 4 WMMAs. M multiple of 64, N multiple of 16,
// K1/K2 multiples of 32. Weights row-major [N][K] (x @ W.T, no transpose).
__global__ void wmma_gemm_bias(const bf16_t* __restrict__ A1, long ldA1, int K1,
                               const bf16_t* __restrict__ W1, long ldW1,
                               const bf16_t* __restrict__ A2, long ldA2, int K2,
                               const bf16_t* __restrict__ W2, long ldW2,
                               const float* __restrict__ bias1,
                               const float* __restrict__ bias2,
                               float* __restrict__ out, long ldOut,
                               int M, int N) {
  int wave = (int)((blockIdx.x * blockDim.x + threadIdx.x) >> 5);
  int lane = (int)(threadIdx.x & 31);
  int ntiles  = N >> 4;
  int mgroups = M >> 6;                 // groups of 4 M-tiles (64 rows)
  if (wave >= ntiles * mgroups) return;
  int nt = wave / mgroups;
  int mg = wave - nt * mgroups;
  int m0 = mg << 6;

  v8f acc0 = {}, acc1 = {}, acc2 = {}, acc3 = {};

  {
    const bf16_t* w = W1 + (size_t)(nt * 16) * (size_t)ldW1;
    const bf16_t* a = A1 + (size_t)m0 * (size_t)ldA1;
    for (int k = 0; k < K1; k += 32) {
      __builtin_prefetch(w + k + 32, 0, 1);   // global_prefetch_b8
      __builtin_prefetch(a + k + 32, 0, 1);
      v16bf bf = load_b_fragT(w, ldW1, k, lane);
      v16bf af0 = load_a_frag(a,                          ldA1, k, lane);
      v16bf af1 = load_a_frag(a + (size_t)16 * ldA1,      ldA1, k, lane);
      v16bf af2 = load_a_frag(a + (size_t)32 * ldA1,      ldA1, k, lane);
      v16bf af3 = load_a_frag(a + (size_t)48 * ldA1,      ldA1, k, lane);
      acc0 = __builtin_amdgcn_wmma_f32_16x16x32_bf16(false, af0, false, bf, (short)0, acc0, false, false);
      acc1 = __builtin_amdgcn_wmma_f32_16x16x32_bf16(false, af1, false, bf, (short)0, acc1, false, false);
      acc2 = __builtin_amdgcn_wmma_f32_16x16x32_bf16(false, af2, false, bf, (short)0, acc2, false, false);
      acc3 = __builtin_amdgcn_wmma_f32_16x16x32_bf16(false, af3, false, bf, (short)0, acc3, false, false);
    }
  }
  if (A2 != nullptr) {
    const bf16_t* w = W2 + (size_t)(nt * 16) * (size_t)ldW2;
    const bf16_t* a = A2 + (size_t)m0 * (size_t)ldA2;
    for (int k = 0; k < K2; k += 32) {
      __builtin_prefetch(w + k + 32, 0, 1);
      __builtin_prefetch(a + k + 32, 0, 1);
      v16bf bf = load_b_fragT(w, ldW2, k, lane);
      v16bf af0 = load_a_frag(a,                          ldA2, k, lane);
      v16bf af1 = load_a_frag(a + (size_t)16 * ldA2,      ldA2, k, lane);
      v16bf af2 = load_a_frag(a + (size_t)32 * ldA2,      ldA2, k, lane);
      v16bf af3 = load_a_frag(a + (size_t)48 * ldA2,      ldA2, k, lane);
      acc0 = __builtin_amdgcn_wmma_f32_16x16x32_bf16(false, af0, false, bf, (short)0, acc0, false, false);
      acc1 = __builtin_amdgcn_wmma_f32_16x16x32_bf16(false, af1, false, bf, (short)0, acc1, false, false);
      acc2 = __builtin_amdgcn_wmma_f32_16x16x32_bf16(false, af2, false, bf, (short)0, acc2, false, false);
      acc3 = __builtin_amdgcn_wmma_f32_16x16x32_bf16(false, af3, false, bf, (short)0, acc3, false, false);
    }
  }

  // C/D layout: VGPR r -> M = tile_base + r + (lane>=16 ? 8 : 0); N = nt*16+lane%16
  int n = nt * 16 + (lane & 15);
  float bb = (bias1 ? bias1[n] : 0.f) + (bias2 ? bias2[n] : 0.f);
  int moff = ((lane & 16) ? 8 : 0);
  v8f accs[4] = {acc0, acc1, acc2, acc3};
#pragma unroll
  for (int mt = 0; mt < 4; ++mt) {
    int mbase = m0 + mt * 16 + moff;
#pragma unroll
    for (int r = 0; r < 8; ++r)
      out[(size_t)(mbase + r) * (size_t)ldOut + n] = accs[mt][r] + bb;
  }
}

// ---------------- elementwise LSTM cell ------------------------------------
__device__ __forceinline__ float sigm(float x) { return 1.f / (1.f + __expf(-x)); }

__global__ void lstm_pointwise(const float* __restrict__ gates,
                               float* __restrict__ h, float* __restrict__ c,
                               bf16_t* __restrict__ h_bf,
                               float* __restrict__ enc_f32,
                               bf16_t* __restrict__ enc_bf,
                               bf16_t* __restrict__ catvec, int t) {
  int idx = (int)(blockIdx.x * blockDim.x + threadIdx.x);
  if (idx >= B_ * H_) return;
  int b = idx / H_, j = idx - b * H_;
  const float* gr = gates + (size_t)b * G4H_;
  float ig = sigm(gr[j]);
  float fg = sigm(gr[H_ + j]);
  float gg = tanhf(gr[2 * H_ + j]);
  float og = sigm(gr[3 * H_ + j]);
  float cn = fg * c[idx] + ig * gg;
  float hn = og * tanhf(cn);
  c[idx] = cn; h[idx] = hn; h_bf[idx] = f2bf(hn);
  if (enc_f32 != nullptr) {
    size_t e = ((size_t)b * S_ + t) * H_ + j;
    enc_f32[e] = hn; enc_bf[e] = f2bf(hn);
  }
  if (catvec != nullptr) catvec[(size_t)b * (2 * H_ + E_) + j] = f2bf(hn);
}

// ---------------- attention energy: e[b,s] = sum_j tanh(pre+hpart)*v -------
__global__ void attn_energy(const float* __restrict__ pre,
                            const float* __restrict__ hpart,
                            const float* __restrict__ v_attn,
                            float* __restrict__ energy) {
  int wave = (int)((blockIdx.x * blockDim.x + threadIdx.x) >> 5);
  int lane = (int)(threadIdx.x & 31);
  if (wave >= B_ * S_) return;
  int b = wave / S_;
  const float* pr = pre + (size_t)wave * H_;
  const float* hp = hpart + (size_t)b * H_;
  float acc = 0.f;
  for (int j = lane; j < H_; j += 32)
    acc += tanhf(pr[j] + hp[j]) * v_attn[j];
#pragma unroll
  for (int off = 16; off > 0; off >>= 1)
    acc += __shfl_down(acc, off, 32);
  if (lane == 0) energy[wave] = acc;
}

// ---------------- softmax over S, context, embed, build GEMM inputs --------
__global__ void softmax_ctx(const float* __restrict__ energy,
                            const float* __restrict__ enc_f32,
                            const int* __restrict__ trg,
                            const float* __restrict__ emb_trg,
                            bf16_t* __restrict__ rnn_in,   // [B][E+H]
                            bf16_t* __restrict__ catvec,   // [B][2H+E]
                            int dstep) {
  __shared__ float sa[S_];
  __shared__ float red[256];
  int b = (int)blockIdx.x, tid = (int)threadIdx.x;
  float e = energy[b * S_ + tid];
  red[tid] = e; __syncthreads();
  for (int s = 128; s > 0; s >>= 1) {
    if (tid < s) red[tid] = fmaxf(red[tid], red[tid + s]);
    __syncthreads();
  }
  float mx = red[0]; __syncthreads();
  float ex = __expf(e - mx);
  red[tid] = ex; __syncthreads();
  for (int s = 128; s > 0; s >>= 1) {
    if (tid < s) red[tid] += red[tid + s];
    __syncthreads();
  }
  float inv = 1.f / red[0];
  sa[tid] = ex * inv;
  __syncthreads();
  // context[b, h] = sum_s attn[s] * enc[b, s, h]
  for (int hh = tid; hh < H_; hh += 256) {
    const float* eb = enc_f32 + (size_t)b * S_ * H_ + hh;
    float acc = 0.f;
    for (int s = 0; s < S_; ++s) acc += sa[s] * eb[(size_t)s * H_];
    bf16_t v = f2bf(acc);
    rnn_in[(size_t)b * (E_ + H_) + E_ + hh] = v;        // [emb, context]
    catvec[(size_t)b * (2 * H_ + E_) + H_ + hh] = v;    // [h, context, emb]
  }
  // teacher forcing: input token at decoder step d is trg[:, d]
  int tok = trg[b * T_ + dstep];
  for (int ee = tid; ee < E_; ee += 256) {
    bf16_t v = f2bf(emb_trg[(size_t)tok * E_ + ee]);
    rnn_in[(size_t)b * (E_ + H_) + ee] = v;
    catvec[(size_t)b * (2 * H_ + E_) + 2 * H_ + ee] = v;
  }
}

// ---------------- small utility kernels ------------------------------------
__global__ void f32_to_bf16_k(const float* __restrict__ in,
                              bf16_t* __restrict__ out, long n) {
  long i = (long)blockIdx.x * blockDim.x + threadIdx.x;
  if (i < n) out[i] = f2bf(in[i]);
}
__global__ void zero_u32_k(unsigned* __restrict__ p, long n) {
  long i = (long)blockIdx.x * blockDim.x + threadIdx.x;
  if (i < n) p[i] = 0u;
}
__global__ void zero_out_row0(float* __restrict__ out) {
  long i = (long)blockIdx.x * blockDim.x + threadIdx.x;
  if (i >= (long)B_ * V_) return;
  long b = i / V_, v = i - b * V_;
  out[(size_t)b * T_ * V_ + v] = 0.f;   // outputs[:, 0, :] = 0
}
__global__ void embed_src_k(const int* __restrict__ src,
                            const float* __restrict__ emb,
                            bf16_t* __restrict__ x) {
  size_t i = (size_t)blockIdx.x * blockDim.x + threadIdx.x;
  if (i >= (size_t)B_ * S_ * E_) return;
  size_t bs = i / E_;
  int e = (int)(i - bs * E_);
  int tok = src[bs];
  x[i] = f2bf(emb[(size_t)tok * E_ + e]);
}

// ---------------------------------------------------------------------------
extern "C" void kernel_launch(void* const* d_in, const int* in_sizes, int n_in,
                              void* d_out, int out_size, void* d_ws, size_t ws_size,
                              hipStream_t stream) {
  const int*   src      = (const int*)d_in[0];
  const int*   trg      = (const int*)d_in[1];
  const float* emb_src  = (const float*)d_in[2];
  const float* emb_trg  = (const float*)d_in[3];
  const float* W_ih_enc = (const float*)d_in[4];
  const float* W_hh_enc = (const float*)d_in[5];
  const float* b_ih_enc = (const float*)d_in[6];
  const float* b_hh_enc = (const float*)d_in[7];
  const float* W_attn   = (const float*)d_in[8];
  const float* b_attn   = (const float*)d_in[9];
  const float* v_attn   = (const float*)d_in[10];
  const float* W_ih_dec = (const float*)d_in[11];
  const float* W_hh_dec = (const float*)d_in[12];
  const float* b_ih_dec = (const float*)d_in[13];
  const float* b_hh_dec = (const float*)d_in[14];
  const float* W_out    = (const float*)d_in[15];
  const float* b_out    = (const float*)d_in[16];
  float* out = (float*)d_out;

  // ---- bump allocator over workspace (~113 MB total) ----
  char* ws = (char*)d_ws;
  auto alloc = [&](size_t bytes) -> void* {
    void* p = (void*)ws;
    ws += (bytes + 255) & ~(size_t)255;
    return p;
  };
  bf16_t* wb_ih_enc = (bf16_t*)alloc((size_t)G4H_ * E_ * 2);
  bf16_t* wb_hh_enc = (bf16_t*)alloc((size_t)G4H_ * H_ * 2);
  bf16_t* wb_attn   = (bf16_t*)alloc((size_t)H_ * 2 * H_ * 2);
  bf16_t* wb_ih_dec = (bf16_t*)alloc((size_t)G4H_ * (E_ + H_) * 2);
  bf16_t* wb_hh_dec = (bf16_t*)alloc((size_t)G4H_ * H_ * 2);
  bf16_t* wb_out    = (bf16_t*)alloc((size_t)V_ * (2 * H_ + E_) * 2);
  bf16_t* x_bf      = (bf16_t*)alloc((size_t)B_ * S_ * E_ * 2);
  float*  enc_f32   = (float*) alloc((size_t)B_ * S_ * H_ * 4);
  bf16_t* enc_bf    = (bf16_t*)alloc((size_t)B_ * S_ * H_ * 2);
  float*  pre       = (float*) alloc((size_t)B_ * S_ * H_ * 4);
  float*  gates     = (float*) alloc((size_t)B_ * G4H_ * 4);
  float*  h         = (float*) alloc((size_t)B_ * H_ * 4);
  float*  c         = (float*) alloc((size_t)B_ * H_ * 4);
  bf16_t* h_bf      = (bf16_t*)alloc((size_t)B_ * H_ * 2);
  float*  hpart     = (float*) alloc((size_t)B_ * H_ * 4);
  float*  energy    = (float*) alloc((size_t)B_ * S_ * 4);
  bf16_t* rnn_in    = (bf16_t*)alloc((size_t)B_ * (E_ + H_) * 2);
  bf16_t* catvec    = (bf16_t*)alloc((size_t)B_ * (2 * H_ + E_) * 2);

  auto cvt = [&](const float* in, bf16_t* o, long n) {
    f32_to_bf16_k<<<(unsigned)((n + 255) / 256), 256, 0, stream>>>(in, o, n);
  };
  auto gemm = [&](const bf16_t* A1, long ldA1, int K1, const bf16_t* W1, long ldW1,
                  const bf16_t* A2, long ldA2, int K2, const bf16_t* W2, long ldW2,
                  const float* b1, const float* b2,
                  float* o, long ldo, int M, int N) {
    int waves = (N >> 4) * (M >> 6);    // 1 wave = 1 N-tile x 4 M-tiles
    int blocks = (waves + 7) / 8;       // 8 wave32 per 256-thread block
    wmma_gemm_bias<<<blocks, 256, 0, stream>>>(A1, ldA1, K1, W1, ldW1,
                                               A2, ldA2, K2, W2, ldW2,
                                               b1, b2, o, ldo, M, N);
  };

  // ---- one-time prep ----
  cvt(W_ih_enc, wb_ih_enc, (long)G4H_ * E_);
  cvt(W_hh_enc, wb_hh_enc, (long)G4H_ * H_);
  cvt(W_attn,   wb_attn,   (long)H_ * 2 * H_);
  cvt(W_ih_dec, wb_ih_dec, (long)G4H_ * (E_ + H_));
  cvt(W_hh_dec, wb_hh_dec, (long)G4H_ * H_);
  cvt(W_out,    wb_out,    (long)V_ * (2 * H_ + E_));
  embed_src_k<<<(unsigned)(((size_t)B_ * S_ * E_ + 255) / 256), 256, 0, stream>>>(
      src, emb_src, x_bf);
  zero_u32_k<<<(B_ * H_ + 255) / 256, 256, 0, stream>>>((unsigned*)h, B_ * H_);
  zero_u32_k<<<(B_ * H_ + 255) / 256, 256, 0, stream>>>((unsigned*)c, B_ * H_);
  zero_u32_k<<<(B_ * H_ / 2 + 255) / 256, 256, 0, stream>>>((unsigned*)h_bf, B_ * H_ / 2);
  zero_out_row0<<<(B_ * V_ + 255) / 256, 256, 0, stream>>>(out);

  // ---- encoder: 256 recurrent steps ----
  for (int t = 0; t < S_; ++t) {
    // gates = x_t @ W_ih.T + h @ W_hh.T + b_ih + b_hh   (M=64, N=2048)
    gemm(x_bf + (size_t)t * E_, (long)S_ * E_, E_, wb_ih_enc, E_,
         h_bf, H_, H_, wb_hh_enc, H_,
         b_ih_enc, b_hh_enc, gates, G4H_, B_, G4H_);
    lstm_pointwise<<<(B_ * H_ + 255) / 256, 256, 0, stream>>>(
        gates, h, c, h_bf, enc_f32, enc_bf, nullptr, t);
  }

  // ---- attention precompute: pre = enc_outs @ W_attn[:, H:].T  (once) ----
  gemm(enc_bf, H_, H_, wb_attn + H_, 2 * H_,
       nullptr, 0, 0, nullptr, 0,
       nullptr, nullptr, pre, H_, B_ * S_, H_);

  // ---- decoder: 255 steps ----
  for (int d = 0; d < T_ - 1; ++d) {
    // hpart = h @ W_attn[:, :H].T + b_attn      (M=64, N=512)
    gemm(h_bf, H_, H_, wb_attn, 2 * H_,
         nullptr, 0, 0, nullptr, 0,
         b_attn, nullptr, hpart, H_, B_, H_);
    attn_energy<<<(B_ * S_ * 32 + 255) / 256, 256, 0, stream>>>(
        pre, hpart, v_attn, energy);
    softmax_ctx<<<B_, 256, 0, stream>>>(energy, enc_f32, trg, emb_trg,
                                        rnn_in, catvec, d);
    // gates = [emb,ctx] @ W_ih_dec.T + h @ W_hh_dec.T + biases  (M=64, N=2048)
    gemm(rnn_in, E_ + H_, E_ + H_, wb_ih_dec, E_ + H_,
         h_bf, H_, H_, wb_hh_dec, H_,
         b_ih_dec, b_hh_dec, gates, G4H_, B_, G4H_);
    lstm_pointwise<<<(B_ * H_ + 255) / 256, 256, 0, stream>>>(
        gates, h, c, h_bf, nullptr, nullptr, catvec, 0);
    // pred = [h,ctx,emb] @ W_out.T + b_out -> out[:, d+1, :]  (M=64, N=4000)
    gemm(catvec, 2 * H_ + E_, 2 * H_ + E_, wb_out, 2 * H_ + E_,
         nullptr, 0, 0, nullptr, 0,
         b_out, nullptr, out + (size_t)(d + 1) * V_, (long)T_ * V_, B_, V_);
  }
}
